// ODEFunc_22204980920985
// MI455X (gfx1250) — matrix operand order repaired and checked by reading, not compiled, not run on hardware
//
#include <hip/hip_runtime.h>
#include <hip/hip_bf16.h>
#include <stdint.h>

typedef __bf16 bf16;
typedef __attribute__((ext_vector_type(16))) bf16  v16bf;
typedef __attribute__((ext_vector_type(8)))  float v8f;
typedef unsigned int u32x4 __attribute__((ext_vector_type(4)));
typedef int          i32x4 __attribute__((ext_vector_type(4)));
typedef int          i32x8 __attribute__((ext_vector_type(8)));

static constexpr int BATCH = 8;
static constexpr int NSEQ  = 2048;
static constexpr int DIM   = 512;
static constexpr float BETA = 1.0f;

#if defined(__has_builtin)
#  if __has_builtin(__builtin_amdgcn_tensor_load_to_lds)
#    define HAVE_TDM 1
#    if __has_include(<hip/amd_detail/amd_gfx1250_TDM.h>)
#      define TDM_ARITY 6   /* amdgpu-toolchain / therock headers */
#    else
#      define TDM_ARITY 5   /* ROCm 7.2 clang-22 */
#    endif
#  endif
#endif

__device__ __forceinline__ v8f wmma_bf16(v16bf a, v16bf b, v8f c) {
  // D = A(16x32 bf16) x B(32x16 bf16) + C(16x16 f32)
  return __builtin_amdgcn_wmma_f32_16x16x32_bf16(
      /*neg_a=*/false, a, /*neg_b=*/false, b,
      /*c_mod=*/(short)0, c, /*reuse_a=*/false, /*reuse_b=*/false);
}

// ---- gfx1250 async copy: global -> LDS, 16B per lane, ASYNCcnt tracked ----
__device__ __forceinline__ void async_g2l_b128(unsigned lds_addr, const void* gaddr) {
  asm volatile("global_load_async_to_lds_b128 %0, %1, off"
               :: "v"(lds_addr), "v"(gaddr) : "memory");
}
__device__ __forceinline__ void wait_async0() {
  asm volatile("s_wait_asynccnt 0" ::: "memory");
}

#if defined(HAVE_TDM)
// ---- Tensor Data Mover: DMA a [rows x 512] bf16 tile of a [2048 x 512]
// row-major tensor into LDS. D# built per ISA 8.3/8.4; issued by one wave,
// completion via TENSORcnt.
__device__ __forceinline__ void tdm_load_rows512(unsigned lds_addr,
                                                 const void* gaddr,
                                                 unsigned rows) {
  const uint64_t ga = (uint64_t)(uintptr_t)gaddr;
  u32x4 g0;
  g0[0] = 1u;                                       // count=1, user descriptor
  g0[1] = lds_addr;                                 // lds_addr (bytes)
  g0[2] = (unsigned)(ga & 0xFFFFFFFFu);             // global_addr[31:0]
  g0[3] = (unsigned)((ga >> 32) & 0x01FFFFFFu)      // global_addr[56:32]
          | (2u << 30);                             // type = 2 ("image")
  i32x8 g1;
  g1[0] = (int)(1u << 16);          // wg_mask=0, data_size=1 (2 bytes)
  g1[1] = (int)(512u << 16);        // tensor_dim0 = 512 (bits[79:48])
  g1[2] = (int)(2048u << 16);       // tensor_dim1 = 2048 (bits[111:80])
  g1[3] = (int)(512u << 16);        // tile_dim0 = 512 (bits[127:112])
  g1[4] = (int)rows;                // tile_dim1 = rows, tile_dim2 = 0
  g1[5] = (int)512;                 // tensor_dim0_stride = 512 (bits[207:160])
  g1[6] = 0;
  g1[7] = 0;
  i32x4 z4 = {0, 0, 0, 0};
#if TDM_ARITY == 6
  i32x8 z8 = {0, 0, 0, 0, 0, 0, 0, 0};
  __builtin_amdgcn_tensor_load_to_lds(g0, g1, z4, z4, z8, 0);
#else
  __builtin_amdgcn_tensor_load_to_lds(g0, g1, z4, z4, 0);
#endif
}
#endif

// Stage key rows [n0, n0+32) x 512 bf16 (32 KB) into LDS at lds_base.
__device__ __forceinline__ void stage_key32(const bf16* kb, int n0,
                                            unsigned lds_base, int tid) {
#if defined(HAVE_TDM)
  if (tid < 32)                       // one wave issues the DMA (EXEC ignored)
    tdm_load_rows512(lds_base, kb + (size_t)n0 * DIM, 32u);
#else
  const char* src = (const char*)(kb + (size_t)n0 * DIM);
#pragma unroll
  for (int i = 0; i < 8; ++i) {
    const int off16 = tid + i * 256;
    async_g2l_b128(lds_base + off16 * 16, src + (size_t)off16 * 16);
  }
#endif
}
__device__ __forceinline__ void wait_stage(int tid) {
#if defined(HAVE_TDM)
  if (tid < 32) __builtin_amdgcn_s_wait_tensorcnt(0);
#else
  wait_async0();
#endif
}

// ---- operand loaders (contiguous K order, same in A and B) -------------------------
// A tile 16x32 (MxK) bf16: lane holds row m=L%16, VGPR i holds
// K = 16*(L/16) + 2i (pairs contiguous).
__device__ __forceinline__ v16bf load_a_bf(const bf16* src, int ld,
                                           int row0, int k0, int lane) {
  const int m = lane & 15, h = lane >> 4;
  const bf16* base = src + (size_t)(row0 + m) * ld + k0;
  v16bf a;
#pragma unroll
  for (int i = 0; i < 8; ++i) {
    const int kk = 16 * h + 2 * i;
    a[2 * i]     = base[kk];
    a[2 * i + 1] = base[kk + 1];
  }
  return a;
}

// Same A layout, fp32 source (global or LDS) converted on the fly.
__device__ __forceinline__ v16bf load_a_f32(const float* src, int ld,
                                            int row0, int k0, int lane) {
  const int m = lane & 15, h = lane >> 4;
  const float* base = src + (size_t)(row0 + m) * ld + k0;
  v16bf a;
#pragma unroll
  for (int i = 0; i < 8; ++i) {
    const int kk = 16 * h + 2 * i;
    a[2 * i]     = (bf16)base[kk];
    a[2 * i + 1] = (bf16)base[kk + 1];
  }
  return a;
}

// B tile 32x16 where B[k][n] = src[(col0+n)*ld + k0+k]  (B = src^T).
// Lane holds col n=L%16; VGPR i holds K = 16*(L/16) + 2i (pairs contiguous).
__device__ __forceinline__ v16bf load_bt_bf(const bf16* src, int ld,
                                            int k0, int col0, int lane) {
  const int n = lane & 15, h = lane >> 4;
  const bf16* base = src + (size_t)(col0 + n) * ld + k0 + 16 * h;
  v16bf b;
#pragma unroll
  for (int i = 0; i < 8; ++i) {
    b[2 * i]     = base[2 * i];
    b[2 * i + 1] = base[2 * i + 1];
  }
  return b;
}

__device__ __forceinline__ v16bf load_bt_f32(const float* __restrict__ src, int ld,
                                             int k0, int col0, int lane) {
  const int n = lane & 15, h = lane >> 4;
  const float* base = src + (size_t)(col0 + n) * ld + k0 + 16 * h;
  v16bf b;
#pragma unroll
  for (int i = 0; i < 8; ++i) {
    b[2 * i]     = (bf16)base[2 * i];
    b[2 * i + 1] = (bf16)base[2 * i + 1];
  }
  return b;
}

// B tile 32x16 where B[k][n] = src[(k0+k)*ld + col0+n] (row-major, strided K).
__device__ __forceinline__ v16bf load_brm_bf(const bf16* src, int ld,
                                             int k0, int col0, int lane) {
  const int n = lane & 15, h = lane >> 4;
  const bf16* base = src + (size_t)(k0 + 16 * h) * ld + col0 + n;
  v16bf b;
#pragma unroll
  for (int i = 0; i < 8; ++i) {
    b[2 * i]     = base[(size_t)(2 * i) * ld];
    b[2 * i + 1] = base[(size_t)(2 * i + 1) * ld];
  }
  return b;
}

// ============ Kernel 1: key_/query_/value_ = x @ W^T (bf16 out) ============
// All 8 waves share one x m-tile -> stage it once (async DMA) instead of 8x.
__global__ void kqv_gemm(const float* __restrict__ x,
                         const float* __restrict__ Wk,
                         const float* __restrict__ Wq,
                         const float* __restrict__ Wv,
                         bf16* __restrict__ keyb,
                         bf16* __restrict__ queryb,
                         bf16* __restrict__ valueb) {
  __shared__ float Xlds[16 * DIM];            // 32 KB x-tile
  const int lane = threadIdx.x & 31, w = threadIdx.x >> 5;
  const int mtile = blockIdx.x;               // 0..127
  const int ctile = blockIdx.y * 8 + w;       // 0..31
  const int bz = blockIdx.z;                  // 0..23
  const int batch = bz / 3, widx = bz % 3;
  const float* W = (widx == 0) ? Wk : (widx == 1) ? Wq : Wv;
  bf16* dst = (widx == 0) ? keyb : (widx == 1) ? queryb : valueb;
  const float* xb = x + (size_t)batch * NSEQ * DIM;
  bf16* db = dst + (size_t)batch * NSEQ * DIM;
  const int row0 = mtile * 16, col0 = ctile * 16;

  {  // stage x rows [row0, row0+16): 16*512*4 B = 2048 x 16B, 8 per thread
    const char* src = (const char*)(xb + (size_t)row0 * DIM);
    const unsigned base = (unsigned)(uintptr_t)Xlds;
#pragma unroll
    for (int i = 0; i < 8; ++i) {
      const int off16 = threadIdx.x + i * 256;
      async_g2l_b128(base + off16 * 16, src + (size_t)off16 * 16);
    }
    wait_async0();
  }
  __syncthreads();

  v8f c = {};
#pragma unroll
  for (int k0 = 0; k0 < DIM; k0 += 32) {
    v16bf a = load_a_f32(Xlds, DIM, 0, k0, lane);    // ds reads + cvt
    v16bf b = load_bt_f32(W, DIM, k0, col0, lane);   // B[k][n] = W[n][k]
    c = wmma_bf16(a, b, c);
  }
  const int mb = row0 + 8 * (lane >> 4);
  const int nn = col0 + (lane & 15);
#pragma unroll
  for (int r = 0; r < 8; ++r)
    db[(size_t)(mb + r) * DIM + nn] = (bf16)c[r];
}

// ====== Kernel 2: row softmax stats of S = query_ @ key_^T (per (b,m)) =====
// Double-buffered key staging: TDM (TENSORcnt) DMA of chunk k+1 overlaps the
// 256 WMMAs of chunk k; all 8 waves share the staged chunk.
__global__ void softmax_stats(const bf16* __restrict__ queryb,
                              const bf16* __restrict__ keyb,
                              float* __restrict__ rowmax,
                              float* __restrict__ rowsum) {
  __shared__ bf16 Klds[2][32 * DIM];          // 2 x 32 KB
  const int lane = threadIdx.x & 31, w = threadIdx.x >> 5;
  const int batch = blockIdx.y;
  const int mtile = blockIdx.x * 8 + w;       // 0..127
  const bf16* qb = queryb + (size_t)batch * NSEQ * DIM;
  const bf16* kb = keyb + (size_t)batch * NSEQ * DIM;
  const int m0 = mtile * 16;
  const int h = lane >> 4;
  const unsigned kbase[2] = {(unsigned)(uintptr_t)&Klds[0][0],
                             (unsigned)(uintptr_t)&Klds[1][0]};

  float rm[8], rs[8];
#pragma unroll
  for (int r = 0; r < 8; ++r) { rm[r] = -1e30f; rs[r] = 0.0f; }

  stage_key32(kb, 0, kbase[0], threadIdx.x);  // prologue DMA

  for (int chunk = 0; chunk < NSEQ / 32; ++chunk) {
    wait_stage(threadIdx.x);
    __syncthreads();    // chunk data visible; previous compute finished
    if (chunk + 1 < NSEQ / 32)
      stage_key32(kb, (chunk + 1) * 32, kbase[(chunk + 1) & 1], threadIdx.x);
    const bf16* kc = &Klds[chunk & 1][0];

#pragma unroll
    for (int nt = 0; nt < 2; ++nt) {
      v8f c = {};
#pragma unroll
      for (int k0 = 0; k0 < DIM; k0 += 32) {
        v16bf a = load_a_bf(qb, DIM, m0, k0, lane);
        v16bf b = load_bt_bf(kc, DIM, k0, nt * 16, lane);  // ds_load_b128
        c = wmma_bf16(a, b, c);
      }
#pragma unroll
      for (int r = 0; r < 8; ++r) {
        const float s = c[r] * BETA;
        float tm = s;   // row max over the 16 n's held across the half-wave
        tm = fmaxf(tm, __shfl_xor(tm, 1, 32));
        tm = fmaxf(tm, __shfl_xor(tm, 2, 32));
        tm = fmaxf(tm, __shfl_xor(tm, 4, 32));
        tm = fmaxf(tm, __shfl_xor(tm, 8, 32));
        const float nm = fmaxf(rm[r], tm);
        float e = __expf(s - nm);
        e += __shfl_xor(e, 1, 32);
        e += __shfl_xor(e, 2, 32);
        e += __shfl_xor(e, 4, 32);
        e += __shfl_xor(e, 8, 32);
        rs[r] = rs[r] * __expf(rm[r] - nm) + e;
        rm[r] = nm;
      }
    }
  }
  if ((lane & 15) == 0) {
#pragma unroll
    for (int r = 0; r < 8; ++r) {
      const int m = m0 + r + 8 * h;
      rowmax[(size_t)batch * NSEQ + m] = rm[r];
      rowsum[(size_t)batch * NSEQ + m] = rs[r];
    }
  }
}

// === Kernel 3: temp = softmax_rows(S) @ value_, fused tangent projection ===
// Value chunk (128 x 512 bf16, 128 KB) staged via ASYNCcnt DMA that overlaps
// the S-tile WMMA chain; strided B gathers then hit LDS, not HBM.
__global__ void attn_pv(const bf16* __restrict__ queryb,
                        const bf16* __restrict__ keyb,
                        const bf16* __restrict__ valueb,
                        const float* __restrict__ rowmax,
                        const float* __restrict__ rowsum,
                        const float* __restrict__ x,
                        float* __restrict__ out) {
  __shared__ bf16 Vlds[128 * DIM];    // 128 KB value chunk
  __shared__ bf16 Plds[16 * 128];     // P chunk: 16 rows x 128 n (bf16)
  __shared__ float projLds[16];

  const int lane = threadIdx.x & 31, w = threadIdx.x >> 5;
  const int batch = blockIdx.y;
  const int m0 = blockIdx.x * 16;
  const int h = lane >> 4, ln = lane & 15;
  const bf16* qb = queryb + (size_t)batch * NSEQ * DIM;
  const bf16* kb = keyb + (size_t)batch * NSEQ * DIM;
  const bf16* vb = valueb + (size_t)batch * NSEQ * DIM;
  const float* xb = x + (size_t)batch * NSEQ * DIM;
  float* ob = out + (size_t)batch * NSEQ * DIM;
  const unsigned vlds_base = (unsigned)(uintptr_t)Vlds;

  float Mr[8], invL[8];
#pragma unroll
  for (int r = 0; r < 8; ++r) {
    const int m = m0 + r + 8 * h;
    Mr[r]   = rowmax[(size_t)batch * NSEQ + m];
    invL[r] = 1.0f / rowsum[(size_t)batch * NSEQ + m];
  }

  v8f acc[4];
#pragma unroll
  for (int t = 0; t < 4; ++t) acc[t] = (v8f){};

  for (int n0 = 0; n0 < NSEQ; n0 += 128) {
    // ---- kick off async DMA of value rows [n0, n0+128): 8192 x 16B --------
    {
      const char* src = (const char*)(vb + (size_t)n0 * DIM);
#pragma unroll
      for (int i = 0; i < 32; ++i) {
        const int off16 = threadIdx.x + i * 256;
        async_g2l_b128(vlds_base + off16 * 16, src + (size_t)off16 * 16);
      }
    }

    // ---- phase 1 (overlapped with DMA): wave w recomputes S tile ----------
    const int nsub = n0 + w * 16;
    v8f c = {};
#pragma unroll
    for (int k0 = 0; k0 < DIM; k0 += 32) {
      v16bf a = load_a_bf(qb, DIM, m0, k0, lane);
      v16bf b = load_bt_bf(kb, DIM, k0, nsub, lane);
      c = wmma_bf16(a, b, c);
    }
#pragma unroll
    for (int r = 0; r < 8; ++r) {
      const float p = __expf(c[r] * BETA - Mr[r]) * invL[r];
      Plds[(r + 8 * h) * 128 + w * 16 + ln] = (bf16)p;
    }
    wait_async0();                    // drain this wave's value DMA
    __syncthreads();                  // all waves' DMA + P writes visible

    // ---- phase 2: wave w does P(16x128) @ V(128x64), cols 64w..64w+63 -----
#pragma unroll
    for (int ksub = 0; ksub < 4; ++ksub) {
      v16bf a;                        // A from LDS in WMMA A layout (ds ops)
      {
        const bf16* base = &Plds[ln * 128 + ksub * 32];
#pragma unroll
        for (int i = 0; i < 8; ++i) {
          const int kk = 16 * h + 2 * i;
          a[2 * i]     = base[kk];
          a[2 * i + 1] = base[kk + 1];
        }
      }
#pragma unroll
      for (int t = 0; t < 4; ++t) {
        v16bf b = load_brm_bf(Vlds, DIM, ksub * 32, w * 64 + t * 16, lane);
        acc[t] = wmma_bf16(a, b, acc[t]);
      }
    }
    __syncthreads();                  // protect Plds/Vlds reuse
  }

  // ---- epilogue: out = temp - (temp . x_row) * x  (per row m) ----
  if (threadIdx.x < 16) projLds[threadIdx.x] = 0.0f;
  __syncthreads();

  float xv[4][8];
#pragma unroll
  for (int t = 0; t < 4; ++t)
#pragma unroll
    for (int r = 0; r < 8; ++r) {
      const int m = m0 + r + 8 * h;
      const int col = w * 64 + t * 16 + ln;
      xv[t][r] = xb[(size_t)m * DIM + col];
    }
#pragma unroll
  for (int r = 0; r < 8; ++r) {
    float pp = 0.0f;
#pragma unroll
    for (int t = 0; t < 4; ++t) pp += acc[t][r] * xv[t][r];
    pp += __shfl_xor(pp, 1, 32);
    pp += __shfl_xor(pp, 2, 32);
    pp += __shfl_xor(pp, 4, 32);
    pp += __shfl_xor(pp, 8, 32);
    if (ln == 0) atomicAdd(&projLds[r + 8 * h], pp);   // ds_add_f32
  }
  __syncthreads();

#pragma unroll
  for (int t = 0; t < 4; ++t)
#pragma unroll
    for (int r = 0; r < 8; ++r) {
      const int m = m0 + r + 8 * h;
      const int col = w * 64 + t * 16 + ln;
      ob[(size_t)m * DIM + col] = acc[t][r] - projLds[r + 8 * h] * xv[t][r];
    }
}

// ============================== launcher ===================================
extern "C" void kernel_launch(void* const* d_in, const int* in_sizes, int n_in,
                              void* d_out, int out_size, void* d_ws, size_t ws_size,
                              hipStream_t stream) {
  // inputs: t (unused), x [8,2048,512] f32, K [512,512], Q [512,512], V [512,512]
  const float* x  = (const float*)d_in[1];
  const float* Wk = (const float*)d_in[2];
  const float* Wq = (const float*)d_in[3];
  const float* Wv = (const float*)d_in[4];
  float* out = (float*)d_out;

  const size_t NBD = (size_t)BATCH * NSEQ * DIM;
  bf16* keyb   = (bf16*)d_ws;
  bf16* queryb = keyb + NBD;
  bf16* valueb = queryb + NBD;
  float* rowmax = (float*)(valueb + NBD);
  float* rowsum = rowmax + (size_t)BATCH * NSEQ;
  // total ws use: 3*NBD*2 bytes (48 MB) + 2*B*N*4 (128 KB)

  kqv_gemm<<<dim3(NSEQ / 16, 4, BATCH * 3), 256, 0, stream>>>(
      x, Wk, Wq, Wv, keyb, queryb, valueb);
  softmax_stats<<<dim3(NSEQ / 16 / 8, BATCH), 256, 0, stream>>>(
      queryb, keyb, rowmax, rowsum);
  attn_pv<<<dim3(NSEQ / 16, BATCH), 256, 0, stream>>>(
      queryb, keyb, valueb, rowmax, rowsum, x, out);
}
